// ReprogrammingLayerForCls_53420803227888
// MI455X (gfx1250) — compile-verified, matrix operand order checked by reading, not run on hardware
//
#include <hip/hip_runtime.h>
#include <stdint.h>

// ---------------------------------------------------------------------------
// Types
// ---------------------------------------------------------------------------
typedef __attribute__((ext_vector_type(16))) __bf16 v16bf;
typedef __attribute__((ext_vector_type(8)))  __bf16 v8bf;
typedef __attribute__((ext_vector_type(8)))  float  v8f;
typedef __attribute__((ext_vector_type(4)))  float  v4f;

union Frag16 { v16bf v; v8bf h[2]; };

__device__ __forceinline__ uint16_t f32_to_bf16_bits(float f) {
  uint32_t x = __builtin_bit_cast(uint32_t, f);
  uint32_t r = x + 0x7FFFu + ((x >> 16) & 1u);              // round-to-nearest-even
  if ((x & 0x7FFFFFFFu) > 0x7F800000u) r = x | 0x00400000u; // quiet NaN
  return (uint16_t)(r >> 16);
}
__device__ __forceinline__ __bf16 f32_to_bf16(float f) {
  uint16_t b = f32_to_bf16_bits(f);
  return __builtin_bit_cast(__bf16, b);
}

// ---------------------------------------------------------------------------
// Problem constants
// ---------------------------------------------------------------------------
#define N_HEADS 8
#define D_KEYS  128
#define D_MODEL 1024
#define D_LLM   4096
#define BQ      32
#define LQ1     513          // L+1 (with cls token)
#define SREAL   1000
#define SPAD    1024
#define MQ      (BQ*LQ1)     // 16416
#define MQP     16512        // 129 * 128 (padded M-grid)

#define LDSTRIDE 48          // bf16 elems per LDS tile row (32 + 16 pad; 96B, 16B-aligned)

// ---------------------------------------------------------------------------
// Elementwise fp32 -> bf16 (4 elems/thread, 8B stores)
// ---------------------------------------------------------------------------
__global__ __launch_bounds__(256)
void cvt_f32_bf16_kernel(const float* __restrict__ src, __bf16* __restrict__ dst)
{
  size_t i4 = ((size_t)blockIdx.x * 256 + threadIdx.x) * 4;
  v4f v = *(const v4f*)(src + i4);
  uint32_t d0 = (uint32_t)f32_to_bf16_bits(v.x) | ((uint32_t)f32_to_bf16_bits(v.y) << 16);
  uint32_t d1 = (uint32_t)f32_to_bf16_bits(v.z) | ((uint32_t)f32_to_bf16_bits(v.w) << 16);
  *(uint64_t*)(dst + i4) = (uint64_t)d0 | ((uint64_t)d1 << 32);
}

// Q input with CLS prepend: Qin[b*513 + l][k] = bf16(l==0 ? cls[k] : tgt[b][l-1][k])
__global__ __launch_bounds__(256)
void cvt_q_input_kernel(const float* __restrict__ tgt, const float* __restrict__ cls,
                        __bf16* __restrict__ dst)
{
  size_t i4 = ((size_t)blockIdx.x * 256 + threadIdx.x) * 4;   // over MQ*1024
  int row = (int)(i4 >> 10);
  int col = (int)(i4 & 1023);
  int b = row / LQ1, l = row - b * LQ1;
  const float* src = (l == 0) ? (cls + col)
                              : (tgt + ((size_t)b * 512 + (l - 1)) * 1024 + col);
  v4f v = *(const v4f*)src;
  uint32_t d0 = (uint32_t)f32_to_bf16_bits(v.x) | ((uint32_t)f32_to_bf16_bits(v.y) << 16);
  uint32_t d1 = (uint32_t)f32_to_bf16_bits(v.z) | ((uint32_t)f32_to_bf16_bits(v.w) << 16);
  *(uint64_t*)(dst + i4) = (uint64_t)d0 | ((uint64_t)d1 << 32);
}

// ---------------------------------------------------------------------------
// One-time weight convert + transpose: Wt[n*K + k] = bf16(W[k*N + n])
// ---------------------------------------------------------------------------
__global__ __launch_bounds__(256)
void wt_transpose_kernel(const float* __restrict__ W, __bf16* __restrict__ Wt,
                         int K, int N)
{
  __shared__ float tile[32][33];
  const int kb = blockIdx.y * 32, nb = blockIdx.x * 32;
  const int tx = threadIdx.x & 31, ty = threadIdx.x >> 5;   // 32 x 8
#pragma unroll
  for (int i = 0; i < 4; ++i) {
    int k = ty + i * 8;
    tile[k][tx] = W[(size_t)(kb + k) * (size_t)N + nb + tx];
  }
  __syncthreads();
#pragma unroll
  for (int i = 0; i < 4; ++i) {
    int n = ty + i * 8;
    Wt[(size_t)(nb + n) * (size_t)K + kb + tx] = f32_to_bf16(tile[tx][n]);
  }
}

// ---------------------------------------------------------------------------
// Tiled GEMM, all-bf16 operands, async-DMA staged, double-buffered LDS:
//   C[M,N] = A[M,K] @ W[K,N];  val = (acc + bias[n]) * outScale
//   A: bf16 [M][K] row-major; Wt: bf16 [N][K] row-major (pre-transposed).
// Block 256 threads (8 waves), tile 128x128, K-step 32.
// Wave grid 4(M) x 2(N); per wave 2 Mtiles x 4 Ntiles of 16x16x32 WMMA.
// Staging: 4x global_load_async_to_lds_b128 per thread per K-step, pipelined
// one tile ahead (s_wait_asynccnt 0x4 keeps the newest tile in flight).
// ---------------------------------------------------------------------------
template<bool OUT_BF16>
__global__ __launch_bounds__(256)
void gemm_bf16_kernel(const __bf16* __restrict__ Abf, const __bf16* __restrict__ Wt,
                      const float* __restrict__ bias, void* __restrict__ Cout,
                      int M, int N, int K, float outScale)
{
  __shared__ __bf16 As[2][128 * LDSTRIDE];
  __shared__ __bf16 Bs[2][128 * LDSTRIDE];

  const int tid  = threadIdx.x;
  const int wid  = tid >> 5;
  const int lane = tid & 31;
  const int l16  = lane & 15;
  const int hlf  = lane >> 4;
  const int waveM = wid & 3;       // rows waveM*32
  const int waveN = wid >> 2;      // cols waveN*64
  const int mBase = blockIdx.y * 128;
  const int nBase = blockIdx.x * 128;

  // Staging addresses (fixed per thread; k0 added at issue time)
  const int nS   = tid & 127;            // B: one n-row, 16 k-elems
  const int kqB  = (tid >> 7) * 16;      // 0 or 16
  const int rowA = tid >> 2;             // A: rows rowA and rowA+64, 8 k-elems
  const int kqA  = (tid & 3) * 8;
  int grA0 = mBase + rowA;      if (grA0 >= M) grA0 = M - 1;
  int grA1 = mBase + rowA + 64; if (grA1 >= M) grA1 = M - 1;
  const __bf16* gB  = Wt  + (size_t)(nBase + nS) * (size_t)K + kqB;
  const __bf16* gA0 = Abf + (size_t)grA0 * (size_t)K + kqA;
  const __bf16* gA1 = Abf + (size_t)grA1 * (size_t)K + kqA;

  auto issue = [&](int k0, int buf) {
    uint32_t laB = (uint32_t)(uintptr_t)(&Bs[buf][nS * LDSTRIDE + kqB]);
    uint64_t gaB = (uint64_t)(uintptr_t)(gB + k0);
    asm volatile("global_load_async_to_lds_b128 %0, %1, off\n\t"
                 "global_load_async_to_lds_b128 %0, %1, off offset:16"
                 :: "v"(laB), "v"(gaB) : "memory");
    uint32_t la0 = (uint32_t)(uintptr_t)(&As[buf][rowA * LDSTRIDE + kqA]);
    uint64_t ga0 = (uint64_t)(uintptr_t)(gA0 + k0);
    asm volatile("global_load_async_to_lds_b128 %0, %1, off"
                 :: "v"(la0), "v"(ga0) : "memory");
    uint32_t la1 = (uint32_t)(uintptr_t)(&As[buf][(rowA + 64) * LDSTRIDE + kqA]);
    uint64_t ga1 = (uint64_t)(uintptr_t)(gA1 + k0);
    asm volatile("global_load_async_to_lds_b128 %0, %1, off"
                 :: "v"(la1), "v"(ga1) : "memory");
  };

  v8f acc[2][4] = {};

  issue(0, 0);
  const int nk = K / 32;
  for (int ki = 0; ki < nk; ++ki) {
    const int buf = ki & 1;
    if (ki + 1 < nk) {
      issue((ki + 1) * 32, buf ^ 1);
      asm volatile("s_wait_asynccnt 0x4" ::: "memory");  // tile ki landed; ki+1 in flight
    } else {
      asm volatile("s_wait_asynccnt 0x0" ::: "memory");
    }
    __syncthreads();

    Frag16 af[2];
#pragma unroll
    for (int mt = 0; mt < 2; ++mt) {
      const __bf16* ap = &As[buf][(waveM * 32 + mt * 16 + l16) * LDSTRIDE];
      af[mt].h[0] = *(const v8bf*)(ap + hlf * 8);        // K = hlf*8 .. +7
      af[mt].h[1] = *(const v8bf*)(ap + 16 + hlf * 8);   // K = 16+hlf*8 .. +7
    }
#pragma unroll
    for (int nt = 0; nt < 4; ++nt) {
      const __bf16* bp2 = &Bs[buf][(waveN * 64 + nt * 16 + l16) * LDSTRIDE];
      Frag16 bfr;
      bfr.h[0] = *(const v8bf*)(bp2 + hlf * 8);
      bfr.h[1] = *(const v8bf*)(bp2 + 16 + hlf * 8);
#pragma unroll
      for (int mt = 0; mt < 2; ++mt)
        acc[mt][nt] = __builtin_amdgcn_wmma_f32_16x16x32_bf16(
            false, af[mt].v, false, bfr.v, (short)0, acc[mt][nt], false, false);
    }
    __syncthreads();
  }

  // ---- epilogue: C row = r + 8*hlf, col = l16 ----
#pragma unroll
  for (int mt = 0; mt < 2; ++mt) {
#pragma unroll
    for (int r = 0; r < 8; ++r) {
      int grow = mBase + waveM * 32 + mt * 16 + r + hlf * 8;
      if (grow >= M) continue;
#pragma unroll
      for (int nt = 0; nt < 4; ++nt) {
        int gcol = nBase + waveN * 64 + nt * 16 + l16;
        float val = (acc[mt][nt][r] + bias[gcol]) * outScale;
        if (OUT_BF16)
          ((__bf16*)Cout)[(size_t)grow * (size_t)N + gcol] = f32_to_bf16(val);
        else
          ((float*)Cout)[(size_t)grow * (size_t)N + gcol] = val;
      }
    }
  }
}

// ---------------------------------------------------------------------------
// V transpose (tiled, coalesced both sides):
//   Vbf [SPAD rows s][1024 he]  ->  Vt [1024 he][SPAD s], zero-fill s >= SREAL
// ---------------------------------------------------------------------------
__global__ __launch_bounds__(256)
void transpose_v_kernel(const __bf16* __restrict__ Vbf, __bf16* __restrict__ Vt)
{
  __shared__ __bf16 tile[32][33];
  const int sb = blockIdx.y * 32, cb = blockIdx.x * 32;
  const int tx = threadIdx.x & 31, ty = threadIdx.x >> 5;   // 32 x 8
  const __bf16 z = f32_to_bf16(0.0f);
#pragma unroll
  for (int i = 0; i < 4; ++i) {
    int s = sb + ty + i * 8;
    tile[ty + i * 8][tx] = (s < SREAL) ? Vbf[((size_t)s << 10) + cb + tx] : z;
  }
  __syncthreads();
#pragma unroll
  for (int i = 0; i < 4; ++i) {
    int c = ty + i * 8;
    Vt[((size_t)(cb + c) << 10) + sb + tx] = tile[tx][c];
  }
}

// ---------------------------------------------------------------------------
// Flash-style attention. Q is pre-scaled by 1/sqrt(E) in its projection.
// Grid: (B*H, ceil(513/64)); block = 128 threads = 4 independent waves.
// ---------------------------------------------------------------------------
__global__ __launch_bounds__(128)
void attn_kernel(const __bf16* __restrict__ Qbf, const __bf16* __restrict__ Kbf,
                 const __bf16* __restrict__ Vt, __bf16* __restrict__ Ctx)
{
  __shared__ __bf16 pLds[4][16][32];   // 1KB per wave, wave-private

  const int tid  = threadIdx.x;
  const int wid  = tid >> 5;
  const int lane = tid & 31;
  const int l16  = lane & 15;
  const int hlf  = lane >> 4;
  const int bh = blockIdx.x;
  const int b = bh >> 3, h = bh & 7;
  const int mbase = blockIdx.y * 64 + wid * 16;

  int lq = mbase + l16; if (lq > LQ1 - 1) lq = LQ1 - 1;
  const __bf16* qrow = Qbf + (((size_t)(b * LQ1 + lq)) << 10) + (h << 7);
  Frag16 qa[4];
#pragma unroll
  for (int kc = 0; kc < 4; ++kc) {
    qa[kc].h[0] = *(const v8bf*)(qrow + kc * 32 + hlf * 8);
    qa[kc].h[1] = *(const v8bf*)(qrow + kc * 32 + 16 + hlf * 8);
  }

  v8f acc[8] = {};
  float mrow[8], lrow[8];
#pragma unroll
  for (int r = 0; r < 8; ++r) { mrow[r] = -__builtin_inff(); lrow[r] = 0.0f; }

  for (int sIter = 0; sIter < SPAD / 32; ++sIter) {
    const int sBase = sIter << 5;
    v8f c0 = {}; v8f c1 = {};
    const __bf16* k0r = Kbf + (((size_t)(sBase + l16)) << 10) + (h << 7);
    const __bf16* k1r = k0r + ((size_t)16 << 10);
#pragma unroll
    for (int kc = 0; kc < 4; ++kc) {
      Frag16 kb0, kb1;
      kb0.h[0] = *(const v8bf*)(k0r + kc * 32 + hlf * 8);
      kb0.h[1] = *(const v8bf*)(k0r + kc * 32 + 16 + hlf * 8);
      kb1.h[0] = *(const v8bf*)(k1r + kc * 32 + hlf * 8);
      kb1.h[1] = *(const v8bf*)(k1r + kc * 32 + 16 + hlf * 8);
      c0 = __builtin_amdgcn_wmma_f32_16x16x32_bf16(false, qa[kc].v, false, kb0.v, (short)0, c0, false, false);
      c1 = __builtin_amdgcn_wmma_f32_16x16x32_bf16(false, qa[kc].v, false, kb1.v, (short)0, c1, false, false);
    }
    const bool v0 = (sBase + l16) < SREAL;
    const bool v1 = (sBase + 16 + l16) < SREAL;
#pragma unroll
    for (int r = 0; r < 8; ++r) {
      float s0 = v0 ? c0[r] : -1e30f;
      float s1 = v1 ? c1[r] : -1e30f;
      float t = fmaxf(s0, s1);
#pragma unroll
      for (int m = 1; m < 16; m <<= 1) t = fmaxf(t, __shfl_xor(t, m, 32));
      float mn   = fmaxf(mrow[r], t);
      float corr = __expf(mrow[r] - mn);
      mrow[r] = mn;
      float p0 = __expf(s0 - mn);
      float p1 = __expf(s1 - mn);
      float rs = p0 + p1;
#pragma unroll
      for (int m = 1; m < 16; m <<= 1) rs += __shfl_xor(rs, m, 32);
      lrow[r] = lrow[r] * corr + rs;
#pragma unroll
      for (int nt = 0; nt < 8; ++nt) acc[nt][r] *= corr;
      c0[r] = p0; c1[r] = p1;
    }
    __bf16* pw = &pLds[wid][0][0];
#pragma unroll
    for (int r = 0; r < 8; ++r) {
      pw[(r + hlf * 8) * 32 + l16]      = f32_to_bf16(c0[r]);
      pw[(r + hlf * 8) * 32 + 16 + l16] = f32_to_bf16(c1[r]);
    }
    asm volatile("s_wait_dscnt 0x0" ::: "memory");
    Frag16 pa;
    pa.h[0] = *(const v8bf*)(pw + l16 * 32 + hlf * 8);
    pa.h[1] = *(const v8bf*)(pw + l16 * 32 + 16 + hlf * 8);
    const __bf16* vtb = Vt + ((size_t)h << 17) + sBase;
#pragma unroll
    for (int nt = 0; nt < 8; ++nt) {
      const __bf16* vr = vtb + ((size_t)(nt * 16 + l16) << 10);
      Frag16 vb;
      vb.h[0] = *(const v8bf*)(vr + hlf * 8);
      vb.h[1] = *(const v8bf*)(vr + 16 + hlf * 8);
      acc[nt] = __builtin_amdgcn_wmma_f32_16x16x32_bf16(false, pa.v, false, vb.v, (short)0, acc[nt], false, false);
    }
  }

#pragma unroll
  for (int r = 0; r < 8; ++r) {
    int lqr = mbase + r + hlf * 8;
    if (lqr >= LQ1) continue;
    float inv = 1.0f / lrow[r];
    __bf16* crow = Ctx + (((size_t)(b * LQ1 + lqr)) << 10) + (h << 7);
#pragma unroll
    for (int nt = 0; nt < 8; ++nt)
      crow[nt * 16 + l16] = f32_to_bf16(acc[nt][r] * inv);
  }
}

// ---------------------------------------------------------------------------
// Launch
// ---------------------------------------------------------------------------
extern "C" void kernel_launch(void* const* d_in, const int* in_sizes, int n_in,
                              void* d_out, int out_size, void* d_ws, size_t ws_size,
                              hipStream_t stream)
{
  (void)in_sizes; (void)n_in; (void)out_size; (void)ws_size;
  const float* tgt  = (const float*)d_in[0];    // [32,512,1024]
  const float* srcE = (const float*)d_in[1];    // [1000,4096]
  const float* valE = (const float*)d_in[2];    // [1000,4096]
  const float* Wq = (const float*)d_in[3];
  const float* bq = (const float*)d_in[4];
  const float* Wk = (const float*)d_in[5];
  const float* bk = (const float*)d_in[6];
  const float* Wv = (const float*)d_in[7];
  const float* bv = (const float*)d_in[8];
  const float* Wo = (const float*)d_in[9];
  const float* bo = (const float*)d_in[10];
  const float* cls = (const float*)d_in[11];
  float* out = (float*)d_out;

  char* ws = (char*)d_ws;
  __bf16* Qin  = (__bf16*)ws;  ws += (size_t)MQ    * 1024 * 2;  // cls-prepended bf16 activations
  __bf16* Sbf  = (__bf16*)ws;  ws += (size_t)SREAL * 4096 * 2;  // source_embedding bf16
  __bf16* Vin  = (__bf16*)ws;  ws += (size_t)SREAL * 4096 * 2;  // value_embedding bf16
  __bf16* Qbf  = (__bf16*)ws;  ws += (size_t)MQ    * 1024 * 2;
  __bf16* Kbf  = (__bf16*)ws;  ws += (size_t)SPAD  * 1024 * 2;
  __bf16* Vbf  = (__bf16*)ws;  ws += (size_t)SPAD  * 1024 * 2;
  __bf16* Vt   = (__bf16*)ws;  ws += (size_t)1024  * SPAD * 2;
  __bf16* Ctx  = (__bf16*)ws;  ws += (size_t)MQ    * 1024 * 2;
  __bf16* WqT  = (__bf16*)ws;  ws += (size_t)1024 * 1024 * 2;   // [n=1024][k=1024]
  __bf16* WkT  = (__bf16*)ws;  ws += (size_t)1024 * 4096 * 2;   // [n=1024][k=4096]
  __bf16* WvT  = (__bf16*)ws;  ws += (size_t)1024 * 4096 * 2;
  __bf16* WoT  = (__bf16*)ws;                                   // [n=4096][k=1024]

  const float qscale = 0.08838834764831845f;   // 1/sqrt(128), folded into Q

  // Activation conversions (one pass each; CLS fused into Q input)
  cvt_q_input_kernel<<<MQ * 1024 / (256 * 4), 256, 0, stream>>>(tgt, cls, Qin);
  cvt_f32_bf16_kernel<<<SREAL * 4096 / (256 * 4), 256, 0, stream>>>(srcE, Sbf);
  cvt_f32_bf16_kernel<<<SREAL * 4096 / (256 * 4), 256, 0, stream>>>(valE, Vin);

  // Weight convert+transpose to bf16 [N][K]
  wt_transpose_kernel<<<dim3(1024 / 32, 1024 / 32), 256, 0, stream>>>(Wq, WqT, 1024, 1024);
  wt_transpose_kernel<<<dim3(1024 / 32, 4096 / 32), 256, 0, stream>>>(Wk, WkT, 4096, 1024);
  wt_transpose_kernel<<<dim3(1024 / 32, 4096 / 32), 256, 0, stream>>>(Wv, WvT, 4096, 1024);
  wt_transpose_kernel<<<dim3(4096 / 32, 1024 / 32), 256, 0, stream>>>(Wo, WoT, 1024, 4096);

  // Projections (all async-DMA pipelined bf16 WMMA GEMMs)
  gemm_bf16_kernel<true><<<dim3(1024 / 128, MQP / 128), 256, 0, stream>>>(
      Qin, WqT, bq, Qbf, MQ, 1024, D_MODEL, qscale);
  gemm_bf16_kernel<true><<<dim3(1024 / 128, SPAD / 128), 256, 0, stream>>>(
      Sbf, WkT, bk, Kbf, SREAL, 1024, D_LLM, 1.0f);
  gemm_bf16_kernel<true><<<dim3(1024 / 128, SPAD / 128), 256, 0, stream>>>(
      Vin, WvT, bv, Vbf, SREAL, 1024, D_LLM, 1.0f);
  // V -> [h][e][s]
  transpose_v_kernel<<<dim3(1024 / 32, SPAD / 32), 256, 0, stream>>>(Vbf, Vt);
  // Fused attention -> bf16 context
  attn_kernel<<<dim3(BQ * N_HEADS, 9), 128, 0, stream>>>(Qbf, Kbf, Vt, Ctx);
  // Output projection: fp32 out + bias
  gemm_bf16_kernel<false><<<dim3(D_LLM / 128, MQP / 128), 256, 0, stream>>>(
      Ctx, WoT, bo, out, MQ, D_LLM, 1024, 1.0f);
}
